// DLightGCN_51144470560839
// MI455X (gfx1250) — compile-verified, matrix operand order checked by reading, not run on hardware
//
#include <hip/hip_runtime.h>

#define D 128

typedef __attribute__((ext_vector_type(16))) _Float16 v16h;
typedef __attribute__((ext_vector_type(8)))  float    v8f;

// ---------------- propagation kernels ----------------

__global__ __launch_bounds__(256) void init_kernel(const float4* __restrict__ uemb,
                                                   const float4* __restrict__ iemb,
                                                   float4* __restrict__ cur,
                                                   float4* __restrict__ acc,
                                                   int n_users4, int total4)
{
    int i = blockIdx.x * 256 + threadIdx.x;
    if (i < total4) {
        float4 v = (i < n_users4) ? uemb[i] : iemb[i - n_users4];
        cur[i] = v;
        acc[i] = v;
    }
}

__global__ __launch_bounds__(256) void zero_kernel(float4* __restrict__ p, int n4) {
    int i = blockIdx.x * 256 + threadIdx.x;
    if (i < n4) p[i] = make_float4(0.f, 0.f, 0.f, 0.f);
}

__global__ __launch_bounds__(256) void accadd_kernel(float4* __restrict__ acc,
                                                     const float4* __restrict__ x, int n4) {
    int i = blockIdx.x * 256 + threadIdx.x;
    if (i < n4) {
        float4 a = acc[i], b = x[i];
        a.x += b.x; a.y += b.y; a.z += b.z; a.w += b.w;
        acc[i] = a;
    }
}

// one wave32 per edge: 32 lanes x float4 = 128 floats = one embedding row
__global__ __launch_bounds__(256) void spmm_kernel(const int* __restrict__ rows,
                                                   const int* __restrict__ cols,
                                                   const float* __restrict__ vals,
                                                   const float* __restrict__ x,
                                                   float* __restrict__ y, int nedges)
{
    int gw = (blockIdx.x * 256 + threadIdx.x) >> 5;   // global wave id (uniform per wave)
    int nw = (gridDim.x * 256) >> 5;
    int l  = threadIdx.x & 31;
    for (int e = gw; e < nedges; e += nw) {
        int r = rows[e], c = cols[e];
        float v = vals[e];
        float4 xv = *(const float4*)(x + (size_t)c * D + l * 4);
        float* yp = y + (size_t)r * D + l * 4;
        atomicAdd(yp + 0, v * xv.x);
        atomicAdd(yp + 1, v * xv.y);
        atomicAdd(yp + 2, v * xv.z);
        atomicAdd(yp + 3, v * xv.w);
    }
}

// ---------------- fused factors + H + score kernel (WMMA) ----------------
// One block = 16 batch rows. 8 waves; wave w owns output columns [w*16, w*16+16).
__global__ __launch_bounds__(256) void factor_score_kernel(
    const int* __restrict__ users, const int* __restrict__ items,
    const float* __restrict__ acc, const float* __restrict__ Wf,
    const float* __restrict__ bfp, const float* __restrict__ Ws,
    float* __restrict__ out, int n_users)
{
    __shared__ _Float16 Eh[2][16][136];      // gathered embeddings (f16), padded stride
    __shared__ float    Ftmp[16][132];       // one factor's pre-norm output (f32)
    __shared__ _Float16 Fh[2][4][16][136];   // normalized factors (f16)
    __shared__ float    Hs[16][16];
    __shared__ float    rnorm[16];

    const int tid  = threadIdx.x;
    const int tile = blockIdx.x;

    // gather ue/ie = acc[node]/4, convert to f16
    for (int idx = tid; idx < 2 * 16 * 128; idx += 256) {
        int side = idx >> 11;
        int r    = (idx >> 7) & 15;
        int d    = idx & 127;
        int b    = tile * 16 + r;
        int node = side ? (n_users + items[b]) : users[b];
        Eh[side][r][d] = (_Float16)(acc[(size_t)node * D + d] * 0.25f);
    }
    __syncthreads();

    const int w    = tid >> 5;          // wave -> N tile
    const int l    = tid & 31;
    const int ln   = l & 15;
    const int hi   = (l >= 16);
    const int k0   = hi ? 8 : 0;        // K sub-offset per lane half (ISA A/B layout)
    const int m0   = hi ? 8 : 0;        // C row offset per lane half
    const int ncol = w * 16 + ln;       // output feature column

    for (int side = 0; side < 2; ++side) {
        for (int k = 0; k < 4; ++k) {
            v8f c = {};
            #pragma unroll
            for (int kc = 0; kc < 4; ++kc) {           // K chunks of 32
                v16h a = {}, bm = {};
                const _Float16* ap = &Eh[side][ln][kc * 32 + k0];
                const float*    bp = Wf + (((size_t)k * D + ncol) * D + kc * 32 + k0);
                #pragma unroll
                for (int j = 0; j < 8; ++j) {
                    a[j]     = ap[j];                  // K = kc*32 + k0 + j
                    a[8 + j] = ap[16 + j];             // K = kc*32 + k0 + 16 + j
                    bm[j]     = (_Float16)bp[j];       // B[k=d][n=f] = Wf[k][f][d]
                    bm[8 + j] = (_Float16)bp[16 + j];
                }
                c = __builtin_amdgcn_wmma_f32_16x16x32_f16(
                        false, a, false, bm, (short)0, c, false, false);
            }
            // bias + relu -> Ftmp (f32)
            float bias = bfp[k * D + ncol];
            #pragma unroll
            for (int i = 0; i < 8; ++i) {
                float v = c[i] + bias;
                Ftmp[m0 + i][ncol] = v > 0.f ? v : 0.f;
            }
            __syncthreads();
            // per-row L2 norm across all 128 columns
            if (tid < 16) {
                float s = 0.f;
                for (int d = 0; d < D; ++d) { float v = Ftmp[tid][d]; s += v * v; }
                rnorm[tid] = 1.f / fmaxf(sqrtf(s), 1e-12f);
            }
            __syncthreads();
            // normalize + stash as f16
            for (int idx = tid; idx < 16 * 128; idx += 256) {
                int r = idx >> 7, d = idx & 127;
                Fh[side][k][r][d] = (_Float16)(Ftmp[r][d] * rnorm[r]);
            }
            __syncthreads();
        }
    }

    // H[b,i,j] = dot(uf[b,i], itf[b,j]); one (b,i,j) per thread
    {
        int b = tid >> 4, p = tid & 15, i = p >> 2, j = p & 3;
        float s = 0.f;
        for (int d = 0; d < D; ++d)
            s += (float)Fh[0][i][b][d] * (float)Fh[1][j][b][d];
        Hs[b][p] = s * Ws[i * 4 + j];
    }
    __syncthreads();
    if (tid < 16) {
        float s = 0.f;
        #pragma unroll
        for (int p = 0; p < 16; ++p) s += Hs[tid][p];
        out[tile * 16 + tid] = s;
    }
}

// ---------------- launch ----------------

extern "C" void kernel_launch(void* const* d_in, const int* in_sizes, int n_in,
                              void* d_out, int out_size, void* d_ws, size_t ws_size,
                              hipStream_t stream)
{
    const int*   users      = (const int*)d_in[0];
    const int*   items      = (const int*)d_in[1];
    const int*   edge_index = (const int*)d_in[2];
    const float* edge_vals  = (const float*)d_in[3];
    const float* user_emb   = (const float*)d_in[4];
    const float* item_emb   = (const float*)d_in[5];
    const float* Wf         = (const float*)d_in[6];
    const float* bfp        = (const float*)d_in[7];
    const float* Ws         = (const float*)d_in[8];
    float*       out        = (float*)d_out;

    const int Bsz = in_sizes[0];
    const int nE  = in_sizes[3];
    const int nU  = in_sizes[4] / D;
    const int nI  = in_sizes[5] / D;
    const int N   = nU + nI;
    const size_t ND = (size_t)N * D;

    float* ws  = (float*)d_ws;
    float* cur = ws;
    float* nxt = ws + ND;
    float* acc = ws + 2 * ND;

    const int n4   = (int)(ND / 4);
    const int blks = (n4 + 255) / 256;

    init_kernel<<<blks, 256, 0, stream>>>((const float4*)user_emb, (const float4*)item_emb,
                                          (float4*)cur, (float4*)acc,
                                          (int)(((size_t)nU * D) / 4), n4);

    const int* rows = edge_index;
    const int* cols = edge_index + nE;
    for (int layer = 0; layer < 3; ++layer) {
        zero_kernel<<<blks, 256, 0, stream>>>((float4*)nxt, n4);
        spmm_kernel<<<16384, 256, 0, stream>>>(rows, cols, edge_vals, cur, nxt, nE);
        accadd_kernel<<<blks, 256, 0, stream>>>((float4*)acc, (const float4*)nxt, n4);
        float* t = cur; cur = nxt; nxt = t;
    }

    factor_score_kernel<<<Bsz / 16, 256, 0, stream>>>(users, items, acc, Wf, bfp, Ws, out, nU);
}
